// PhaseAssociativeLayer_89060441850280
// MI455X (gfx1250) — compile-verified, matrix operand order checked by reading, not run on hardware
//
#include <hip/hip_runtime.h>
#include <hip/hip_bf16.h>

typedef __attribute__((ext_vector_type(16))) __bf16 v16bf;
typedef __attribute__((ext_vector_type(8)))  float  v8f;
typedef __bf16 bf16_t;

#define TLEN 2048
#define DIMC 1024
#define NH   16
#define HDIM 64

union Frag {
    v16bf v;
    uint4 q[2];
};

__device__ inline v8f wmma_bf16(v16bf a, v16bf b, v8f c) {
    return __builtin_amdgcn_wmma_f32_16x16x32_bf16(false, a, false, b, (short)0, c, false, false);
}

// ---------------------------------------------------------------------------
// Generic bf16 GEMM: C(MxN) = A(MxK) * B(NxK)^T, fp32 accumulate via WMMA.
// Block = 128 threads (4 waves), 128x128 block tile, 64x64 per wave:
// 16 WMMAs per 16 global_load_b128 per K-step (32 FLOP/byte from cache).
// ---------------------------------------------------------------------------
__global__ __launch_bounds__(128) void gemm_bf16_kernel(
    const bf16_t* __restrict__ A, const bf16_t* __restrict__ B,
    float* __restrict__ C, int M, int N, int K)
{
    const int lane = threadIdx.x & 31;
    const int wave = threadIdx.x >> 5;
    const int r  = lane & 15;   // fragment row
    const int hk = lane >> 4;   // K-half select
    const int wm = blockIdx.y * 128 + (wave >> 1) * 64;
    const int wn = blockIdx.x * 128 + (wave & 1) * 64;

    v8f acc[4][4];
#pragma unroll
    for (int i = 0; i < 4; ++i)
#pragma unroll
        for (int j = 0; j < 4; ++j)
#pragma unroll
            for (int e = 0; e < 8; ++e) acc[i][j][e] = 0.0f;

    for (int k0 = 0; k0 < K; k0 += 32) {
        Frag a[4], b[4];
#pragma unroll
        for (int i = 0; i < 4; ++i) {
            const bf16_t* pa = A + (size_t)(wm + i * 16 + r) * K + k0 + hk * 8;
            a[i].q[0] = *(const uint4*)pa;
            a[i].q[1] = *(const uint4*)(pa + 16);
            const bf16_t* pb = B + (size_t)(wn + i * 16 + r) * K + k0 + hk * 8;
            b[i].q[0] = *(const uint4*)pb;
            b[i].q[1] = *(const uint4*)(pb + 16);
            if (k0 + 64 < K) {
                __builtin_prefetch(pa + 64, 0, 3);
                __builtin_prefetch(pb + 64, 0, 3);
            }
        }
#pragma unroll
        for (int i = 0; i < 4; ++i)
#pragma unroll
            for (int j = 0; j < 4; ++j)
                acc[i][j] = wmma_bf16(a[i].v, b[j].v, acc[i][j]);
    }

#pragma unroll
    for (int i = 0; i < 4; ++i)
#pragma unroll
        for (int j = 0; j < 4; ++j)
#pragma unroll
            for (int rr = 0; rr < 8; ++rr) {
                const int m = wm + i * 16 + rr + 8 * hk;
                const int n = wn + j * 16 + r;
                C[(size_t)m * N + n] = acc[i][j][rr];
            }
}

// ---------------------------------------------------------------------------
// Split x (T,DIM,2) into bf16 xcat (T x 2*DIM) = [xr | xi]
// ---------------------------------------------------------------------------
__global__ void prep_x_kernel(const float* __restrict__ x, bf16_t* __restrict__ xcat)
{
    const int idx = blockIdx.x * blockDim.x + threadIdx.x;  // t*1024 + d
    const int t = idx >> 10, d = idx & 1023;
    const float xr = x[((size_t)t * DIMC + d) * 2];
    const float xi = x[((size_t)t * DIMC + d) * 2 + 1];
    xcat[(size_t)t * 2048 + d]        = (bf16_t)xr;
    xcat[(size_t)t * 2048 + 1024 + d] = (bf16_t)xi;
}

// ---------------------------------------------------------------------------
// Build Bqkv (6144 x 2048): rows [0,3072) = [wr | -wi] (real out),
//                            rows [3072,6144) = [wi | wr] (imag out)
// ---------------------------------------------------------------------------
__global__ void prep_wqkv_kernel(const float* __restrict__ wr, const float* __restrict__ wi,
                                 bf16_t* __restrict__ Bq)
{
    const int idx = blockIdx.x * blockDim.x + threadIdx.x;  // n*1024 + k, n<3072
    const int n = idx >> 10, k = idx & 1023;
    const float a = wr[(size_t)n * 1024 + k];
    const float b = wi[(size_t)n * 1024 + k];
    Bq[(size_t)n * 2048 + k]                   = (bf16_t)a;
    Bq[(size_t)n * 2048 + 1024 + k]            = (bf16_t)(-b);
    Bq[((size_t)(3072 + n)) * 2048 + k]        = (bf16_t)b;
    Bq[((size_t)(3072 + n)) * 2048 + 1024 + k] = (bf16_t)a;
}

// ---------------------------------------------------------------------------
// Build Bout (2048 x 2048): rows [0,1024)=[o_wr|-o_wi], rows [1024,2048)=[o_wi|o_wr]
// ---------------------------------------------------------------------------
__global__ void prep_wo_kernel(const float* __restrict__ wr, const float* __restrict__ wi,
                               bf16_t* __restrict__ Bo)
{
    const int idx = blockIdx.x * blockDim.x + threadIdx.x;  // n*1024 + k, n<1024
    const int n = idx >> 10, k = idx & 1023;
    const float a = wr[(size_t)n * 1024 + k];
    const float b = wi[(size_t)n * 1024 + k];
    Bo[(size_t)n * 2048 + k]                   = (bf16_t)a;
    Bo[(size_t)n * 2048 + 1024 + k]            = (bf16_t)(-b);
    Bo[((size_t)(1024 + n)) * 2048 + k]        = (bf16_t)b;
    Bo[((size_t)(1024 + n)) * 2048 + 1024 + k] = (bf16_t)a;
}

// ---------------------------------------------------------------------------
// dt / p gates: one thread per (h,t); dot over DIM.
// Stores p[h][t] and neglog[h][t] = -log(gamma + 1e-6)
// ---------------------------------------------------------------------------
__global__ void dtp_kernel(const float* __restrict__ x,
                           const float* __restrict__ dt_w, const float* __restrict__ dt_b,
                           const float* __restrict__ dt_bias,
                           const float* __restrict__ pg_w, const float* __restrict__ pg_b,
                           float* __restrict__ p_out, float* __restrict__ neglog)
{
    const int idx = blockIdx.x * blockDim.x + threadIdx.x;  // h*T + t
    if (idx >= NH * TLEN) return;
    const int h = idx / TLEN, t = idx % TLEN;
    const float* xt = x + (size_t)t * DIMC * 2;
    const float* dwr = dt_w + (size_t)h * 2048;
    const float* dwi = dwr + 1024;
    const float* pw  = pg_w + (size_t)h * 1024;
    float accd = 0.f, accp = 0.f;
    for (int d = 0; d < DIMC; ++d) {
        const float vr = xt[2 * d], vi = xt[2 * d + 1];
        accd += vr * dwr[d] + vi * dwi[d];
        accp += sqrtf(vr * vr + vi * vi + 1e-8f) * pw[d];
    }
    const float z = accd + dt_b[h] + dt_bias[h];
    const float dt = (z > 20.f) ? z : log1pf(__expf(z));
    const float pv = 1.f / (1.f + __expf(-(accp + pg_b[h])));
    const float gamma = __expf(-dt) * (1.f - pv) + pv;
    p_out[idx]  = pv;
    neglog[idx] = -logf(gamma + 1e-6f);
}

// ---------------------------------------------------------------------------
// Per-head inclusive scan of neglog -> C, plus D_last[t] = exp(C[t]-C[T-1])
// One block (256 threads) per head; 8 elements per thread (T = 2048).
// ---------------------------------------------------------------------------
__global__ __launch_bounds__(256) void scan_kernel(const float* __restrict__ neglog,
                                                   float* __restrict__ Cw, float* __restrict__ Dl)
{
    const int h = blockIdx.x;
    __shared__ float part[256];
    const float* in = neglog + (size_t)h * TLEN;
    float* out = Cw + (size_t)h * TLEN;
    float* dl  = Dl + (size_t)h * TLEN;
    const int tid = threadIdx.x;
    float v[8], s = 0.f;
#pragma unroll
    for (int e = 0; e < 8; ++e) { v[e] = in[tid * 8 + e]; s += v[e]; }
    part[tid] = s;
    __syncthreads();
    for (int off = 1; off < 256; off <<= 1) {
        const float add = (tid >= off) ? part[tid - off] : 0.f;
        __syncthreads();
        part[tid] += add;
        __syncthreads();
    }
    const float prefix = (tid > 0) ? part[tid - 1] : 0.f;
    const float clast  = part[255];
    float run = prefix;
#pragma unroll
    for (int e = 0; e < 8; ++e) {
        run += v[e];
        out[tid * 8 + e] = run;
        dl[tid * 8 + e]  = __expf(run - clast);
    }
}

// ---------------------------------------------------------------------------
// Pack qkv fp32 GEMM result -> per-head bf16 operands with RoPE, scale, vp.
// qkvbuf: T x 6144 with columns [qr kr vr (3072) | qi ki vi (3072)]
// ---------------------------------------------------------------------------
__global__ void pack_qkv_kernel(const float* __restrict__ qkvbuf, const float* __restrict__ p_arr,
                                bf16_t* __restrict__ Qcat, bf16_t* __restrict__ Qrot,
                                bf16_t* __restrict__ Kcat,
                                bf16_t* __restrict__ vprT, bf16_t* __restrict__ vpiT,
                                bf16_t* __restrict__ vpiTn)
{
    const int idx = blockIdx.x * blockDim.x + threadIdx.x;  // ((h*T)+t)*64 + d
    const int d = idx & 63;
    const int t = (idx >> 6) & (TLEN - 1);
    const int h = idx >> 17;  // 64*2048 = 2^17
    const size_t row = (size_t)t * 6144;
    const float qr = qkvbuf[row + h * 64 + d];
    const float kr = qkvbuf[row + 1024 + h * 64 + d];
    const float vr = qkvbuf[row + 2048 + h * 64 + d];
    const float qi = qkvbuf[row + 3072 + h * 64 + d];
    const float ki = qkvbuf[row + 3072 + 1024 + h * 64 + d];
    const float vi = qkvbuf[row + 3072 + 2048 + h * 64 + d];
    // RoPE: freq = 10000^(-d/64)
    const float fr  = __powf(10000.f, -(float)d * (1.0f / 64.0f));
    const float ang = (float)t * fr;
    const float cs = __cosf(ang), sn = __sinf(ang);
    const float qr2 = qr * cs - qi * sn, qi2 = qr * sn + qi * cs;
    const float kr2 = kr * cs - ki * sn, ki2 = kr * sn + ki * cs;
    const float scale = 0.125f;  // HD^-0.5
    const size_t qb = ((size_t)h * TLEN + t) * 128;
    Qcat[qb + d]      = (bf16_t)(qr2 * scale);
    Qcat[qb + 64 + d] = (bf16_t)(qi2 * scale);
    Qrot[qb + d]      = (bf16_t)(qi2 * scale);
    Qrot[qb + 64 + d] = (bf16_t)(-qr2 * scale);
    Kcat[qb + d]      = (bf16_t)kr2;
    Kcat[qb + 64 + d] = (bf16_t)ki2;
    const float pv = p_arr[h * TLEN + t];
    const float om = 1.f - pv;
    vprT[((size_t)h * 64 + d) * TLEN + t]  = (bf16_t)(vr * om);
    vpiT[((size_t)h * 64 + d) * TLEN + t]  = (bf16_t)(vi * om);
    vpiTn[((size_t)h * 64 + d) * TLEN + t] = (bf16_t)(-vi * om);
}

// ---------------------------------------------------------------------------
// Flash-style decayed complex attention. Grid: (T/64, H). 128 threads = 4 waves.
// Each wave owns 16 output rows; streams 32-wide column blocks.
// ---------------------------------------------------------------------------
__global__ __launch_bounds__(128) void attn_kernel(
    const bf16_t* __restrict__ Qcat, const bf16_t* __restrict__ Qrot,
    const bf16_t* __restrict__ Kcat,
    const bf16_t* __restrict__ vprT, const bf16_t* __restrict__ vpiT,
    const bf16_t* __restrict__ vpiTn,
    const float* __restrict__ Cw,
    bf16_t* __restrict__ ycat)
{
    const int h  = blockIdx.y;
    const int i0 = blockIdx.x * 64;
    const int lane = threadIdx.x & 31;
    const int wave = threadIdx.x >> 5;
    const int r  = lane & 15;
    const int hk = lane >> 4;
    const int iw = i0 + wave * 16;

    const bf16_t* Qc = Qcat + (size_t)h * TLEN * 128;
    const bf16_t* Qr = Qrot + (size_t)h * TLEN * 128;
    const bf16_t* Kc = Kcat + (size_t)h * TLEN * 128;
    const bf16_t* Vr = vprT + (size_t)h * 64 * TLEN;
    const bf16_t* Vi = vpiT + (size_t)h * 64 * TLEN;
    const bf16_t* Vn = vpiTn + (size_t)h * 64 * TLEN;
    const float*  Ch = Cw + (size_t)h * TLEN;

    // Q fragments (16 rows x K=128 -> 4 k-steps), both variants
    Frag aq[4], ax[4];
#pragma unroll
    for (int kk = 0; kk < 4; ++kk) {
        const bf16_t* p = Qc + (size_t)(iw + r) * 128 + kk * 32 + hk * 8;
        aq[kk].q[0] = *(const uint4*)p;
        aq[kk].q[1] = *(const uint4*)(p + 16);
        const bf16_t* p2 = Qr + (size_t)(iw + r) * 128 + kk * 32 + hk * 8;
        ax[kk].q[0] = *(const uint4*)p2;
        ax[kk].q[1] = *(const uint4*)(p2 + 16);
    }
    float ci[8];
#pragma unroll
    for (int rr = 0; rr < 8; ++rr) ci[rr] = Ch[iw + rr + 8 * hk];

    v8f yr[4], yi[4];
#pragma unroll
    for (int nf = 0; nf < 4; ++nf)
#pragma unroll
        for (int e = 0; e < 8; ++e) { yr[nf][e] = 0.f; yi[nf][e] = 0.f; }

    __shared__ bf16_t lds_a[4][2][16][32];  // [wave][re/im][m][j]

    const int jend = i0 + 64;
    for (int j0 = 0; j0 < jend; j0 += 32) {
        // stage 1: w = Qcat @ Kcat^T (wr) and Qrot @ Kcat^T (wi)
        v8f wr[2], wi[2];
#pragma unroll
        for (int jf = 0; jf < 2; ++jf)
#pragma unroll
            for (int e = 0; e < 8; ++e) { wr[jf][e] = 0.f; wi[jf][e] = 0.f; }
#pragma unroll
        for (int jf = 0; jf < 2; ++jf) {
#pragma unroll
            for (int kk = 0; kk < 4; ++kk) {
                Frag b;
                const bf16_t* p = Kc + (size_t)(j0 + jf * 16 + r) * 128 + kk * 32 + hk * 8;
                b.q[0] = *(const uint4*)p;
                b.q[1] = *(const uint4*)(p + 16);
                wr[jf] = wmma_bf16(aq[kk].v, b.v, wr[jf]);
                wi[jf] = wmma_bf16(ax[kk].v, b.v, wi[jf]);
            }
        }
        // decay + causal mask, write masked tiles to LDS in bf16
#pragma unroll
        for (int jf = 0; jf < 2; ++jf) {
            const int j = j0 + jf * 16 + r;  // n index of C/D layout
            const float cj = Ch[j];
#pragma unroll
            for (int rr = 0; rr < 8; ++rr) {
                const int i = iw + rr + 8 * hk;
                const int m = rr + 8 * hk;
                const float dm = (j <= i) ? __expf(fminf(cj - ci[rr], 0.f)) : 0.f;
                lds_a[wave][0][m][jf * 16 + r] = (bf16_t)(wr[jf][rr] * dm);
                lds_a[wave][1][m][jf * 16 + r] = (bf16_t)(wi[jf][rr] * dm);
            }
        }
        __syncthreads();
        // reload masked tiles as A fragments (16 x 32)
        Frag fa, fb;
        {
            const bf16_t* p = &lds_a[wave][0][r][hk * 8];
            fa.q[0] = *(const uint4*)p;
            fa.q[1] = *(const uint4*)(p + 16);
            const bf16_t* p2 = &lds_a[wave][1][r][hk * 8];
            fb.q[0] = *(const uint4*)p2;
            fb.q[1] = *(const uint4*)(p2 + 16);
        }
        // stage 2: yr += ar@vpr + ai@(-vpi); yi += ar@vpi + ai@vpr
#pragma unroll
        for (int nf = 0; nf < 4; ++nf) {
            Frag br, bi, bn;
            const bf16_t* pr = Vr + (size_t)(nf * 16 + r) * TLEN + j0 + hk * 8;
            br.q[0] = *(const uint4*)pr; br.q[1] = *(const uint4*)(pr + 16);
            const bf16_t* pi = Vi + (size_t)(nf * 16 + r) * TLEN + j0 + hk * 8;
            bi.q[0] = *(const uint4*)pi; bi.q[1] = *(const uint4*)(pi + 16);
            const bf16_t* pn = Vn + (size_t)(nf * 16 + r) * TLEN + j0 + hk * 8;
            bn.q[0] = *(const uint4*)pn; bn.q[1] = *(const uint4*)(pn + 16);
            yr[nf] = wmma_bf16(fa.v, br.v, yr[nf]);
            yr[nf] = wmma_bf16(fb.v, bn.v, yr[nf]);
            yi[nf] = wmma_bf16(fa.v, bi.v, yi[nf]);
            yi[nf] = wmma_bf16(fb.v, br.v, yi[nf]);
        }
        __syncthreads();
    }
    // epilogue: ycat[t][h*64+d] = yr, ycat[t][1024+h*64+d] = yi (bf16)
#pragma unroll
    for (int nf = 0; nf < 4; ++nf)
#pragma unroll
        for (int rr = 0; rr < 8; ++rr) {
            const int tt = iw + rr + 8 * hk;
            const int d  = nf * 16 + r;
            ycat[(size_t)tt * 2048 + h * 64 + d]        = (bf16_t)yr[nf][rr];
            ycat[(size_t)tt * 2048 + 1024 + h * 64 + d] = (bf16_t)yi[nf][rr];
        }
}

// ---------------------------------------------------------------------------
// new_state: per head 64x64 complex reduction over T. One thread per (h,i,j).
// ---------------------------------------------------------------------------
__global__ __launch_bounds__(256) void state_kernel(
    const bf16_t* __restrict__ vprT, const bf16_t* __restrict__ vpiT,
    const bf16_t* __restrict__ Kcat, const float* __restrict__ Dl,
    float* __restrict__ out_state)
{
    const int h = blockIdx.x;
    const int idx = blockIdx.y * blockDim.x + threadIdx.x;  // 0..4095
    const int i = idx >> 6, j = idx & 63;
    const bf16_t* vr = vprT + ((size_t)h * 64 + i) * TLEN;
    const bf16_t* vi = vpiT + ((size_t)h * 64 + i) * TLEN;
    const bf16_t* kc = Kcat + (size_t)h * TLEN * 128;
    const float* dl = Dl + (size_t)h * TLEN;
    float sr = 0.f, si = 0.f;
    for (int t = 0; t < TLEN; ++t) {
        const float d  = dl[t];
        const float wvr = (float)vr[t] * d;
        const float wvi = (float)vi[t] * d;
        const float krt = (float)kc[(size_t)t * 128 + j];
        const float kit = (float)kc[(size_t)t * 128 + 64 + j];
        sr += wvr * krt + wvi * kit;
        si += wvi * krt - wvr * kit;
    }
    const size_t o = (((size_t)h * 64 + i) * 64 + j) * 2;
    out_state[o]     = sr;
    out_state[o + 1] = si;
}

// ---------------------------------------------------------------------------
// Interleave final GEMM result (T x 2048 = [re|im]) into (T, DIM, 2)
// ---------------------------------------------------------------------------
__global__ void pack_out_kernel(const float* __restrict__ outcat, float* __restrict__ out)
{
    const int idx = blockIdx.x * blockDim.x + threadIdx.x;  // t*1024 + d
    const int t = idx >> 10, d = idx & 1023;
    out[((size_t)t * DIMC + d) * 2]     = outcat[(size_t)t * 2048 + d];
    out[((size_t)t * DIMC + d) * 2 + 1] = outcat[(size_t)t * 2048 + 1024 + d];
}

// ---------------------------------------------------------------------------
extern "C" void kernel_launch(void* const* d_in, const int* in_sizes, int n_in,
                              void* d_out, int out_size, void* d_ws, size_t ws_size,
                              hipStream_t stream)
{
    const float* x       = (const float*)d_in[0];
    const float* qkv_wr  = (const float*)d_in[1];
    const float* qkv_wi  = (const float*)d_in[2];
    const float* o_wr    = (const float*)d_in[3];
    const float* o_wi    = (const float*)d_in[4];
    const float* dt_w    = (const float*)d_in[5];
    const float* dt_b    = (const float*)d_in[6];
    const float* dt_bias = (const float*)d_in[7];
    const float* pg_w    = (const float*)d_in[8];
    const float* pg_b    = (const float*)d_in[9];
    float* out = (float*)d_out;
    float* out_state = out + (size_t)TLEN * DIMC * 2;

    size_t off = 0;
    auto alloc = [&](size_t bytes) -> void* {
        void* p = (char*)d_ws + off;
        off += (bytes + 255) & ~(size_t)255;
        return p;
    };
    bf16_t* xcat   = (bf16_t*)alloc((size_t)TLEN * 2048 * 2);
    bf16_t* Bqkv   = (bf16_t*)alloc((size_t)6144 * 2048 * 2);
    float*  qkvbuf = (float*) alloc((size_t)TLEN * 6144 * 4);
    bf16_t* Qcat   = (bf16_t*)alloc((size_t)NH * TLEN * 128 * 2);
    bf16_t* Qrot   = (bf16_t*)alloc((size_t)NH * TLEN * 128 * 2);
    bf16_t* Kcat   = (bf16_t*)alloc((size_t)NH * TLEN * 128 * 2);
    bf16_t* vprT   = (bf16_t*)alloc((size_t)NH * 64 * TLEN * 2);
    bf16_t* vpiT   = (bf16_t*)alloc((size_t)NH * 64 * TLEN * 2);
    bf16_t* vpiTn  = (bf16_t*)alloc((size_t)NH * 64 * TLEN * 2);
    float*  p_arr  = (float*) alloc((size_t)NH * TLEN * 4);
    float*  neglog = (float*) alloc((size_t)NH * TLEN * 4);
    float*  Cw     = (float*) alloc((size_t)NH * TLEN * 4);
    float*  Dl     = (float*) alloc((size_t)NH * TLEN * 4);
    bf16_t* ycat   = (bf16_t*)alloc((size_t)TLEN * 2048 * 2);
    bf16_t* Bout   = (bf16_t*)alloc((size_t)2048 * 2048 * 2);
    float*  outcat = (float*) alloc((size_t)TLEN * 2048 * 4);
    if (off > ws_size) return;  // insufficient scratch

    // 1) operand prep
    prep_x_kernel<<<(TLEN * DIMC) / 256, 256, 0, stream>>>(x, xcat);
    prep_wqkv_kernel<<<(3072 * 1024) / 256, 256, 0, stream>>>(qkv_wr, qkv_wi, Bqkv);
    prep_wo_kernel<<<(1024 * 1024) / 256, 256, 0, stream>>>(o_wr, o_wi, Bout);
    // 2) gates + scan
    dtp_kernel<<<(NH * TLEN) / 256, 256, 0, stream>>>(x, dt_w, dt_b, dt_bias, pg_w, pg_b,
                                                      p_arr, neglog);
    scan_kernel<<<NH, 256, 0, stream>>>(neglog, Cw, Dl);
    // 3) QKV projection (WMMA): 2048 x 6144 x 2048
    gemm_bf16_kernel<<<dim3(6144 / 128, TLEN / 128), 128, 0, stream>>>(
        xcat, Bqkv, qkvbuf, TLEN, 6144, 2048);
    // 4) RoPE + pack per-head bf16 operands
    pack_qkv_kernel<<<(NH * TLEN * 64) / 256, 256, 0, stream>>>(
        qkvbuf, p_arr, Qcat, Qrot, Kcat, vprT, vpiT, vpiTn);
    // 5) decayed complex attention (WMMA, flash-style)
    attn_kernel<<<dim3(TLEN / 64, NH), 128, 0, stream>>>(
        Qcat, Qrot, Kcat, vprT, vpiT, vpiTn, Cw, ycat);
    // 6) new_state
    state_kernel<<<dim3(NH, 16), 256, 0, stream>>>(vprT, vpiT, Kcat, Dl, out_state);
    // 7) output projection (WMMA): 2048 x 2048 x 2048
    gemm_bf16_kernel<<<dim3(2048 / 128, TLEN / 128), 128, 0, stream>>>(
        ycat, Bout, outcat, TLEN, 2048, 2048);
    // 8) interleave to (T, DIM, 2)
    pack_out_kernel<<<(TLEN * DIMC) / 256, 256, 0, stream>>>(outcat, out);
}
